// SS2D_3298534883723
// MI455X (gfx1250) — matrix operand order repaired and checked by reading, not compiled, not run on hardware
//
#include <hip/hip_runtime.h>
#include <hip/hip_bf16.h>
#include <math.h>
#include <stdint.h>

// ---- problem constants ----
#define DM      96
#define DI      48
#define DSTATE  16
#define RNK     6
#define NK      4
#define BB      4
#define HH      128
#define WW      128
#define LL      (HH*WW)      // 16384
#define NC      128          // number of scan chunks
#define CH      128          // chunk length (NC*CH == LL)
#define EM      80           // fused projection rows: 48 delta + 16 B + 16 C
#define EKP     64           // padded K for f16 staging

typedef _Float16 half_t;
typedef __attribute__((ext_vector_type(16))) _Float16     v16h;
typedef __attribute__((ext_vector_type(8)))  float        v8f;
typedef __attribute__((ext_vector_type(4)))  unsigned int v4u;
typedef __attribute__((ext_vector_type(8)))  int          v8i;
typedef __attribute__((ext_vector_type(4)))  int          v4i;

// ---------------- WMMA helpers (CDNA5 16x16x32 f16 -> f32) ----------------
__device__ __forceinline__ v8f wmma_f16(v16h a, v16h b, v8f c) {
    // 8 args: (neg_a, A, neg_b, B, c_mod, C, reuse_a, reuse_b)
    return __builtin_amdgcn_wmma_f32_16x16x32_f16(false, a, false, b,
                                                  (short)0, c, false, false);
}

// A fragment: 16x32 f16, row-major tile in LDS (f16), stride `ld` halves.
// lanes 0-15 hold K = 0..7,16..23 ; lanes 16-31 hold K = 8..15,24..31
__device__ __forceinline__ v16h load_a_frag(const half_t* tile, int ld) {
    int lane = threadIdx.x & 31;
    const half_t* r0 = tile + (lane & 15) * ld + ((lane < 16) ? 0 : 8);
    v16h a;
#pragma unroll
    for (int e = 0; e < 8; ++e) a[e] = r0[e];
#pragma unroll
    for (int e = 0; e < 8; ++e) a[8 + e] = r0[16 + e];
    return a;
}

// A fragment from an f32 tile in LDS (convert to f16 in registers)
__device__ __forceinline__ v16h load_a_frag_f32(const float* tile, int ld) {
    int lane = threadIdx.x & 31;
    const float* r0 = tile + (lane & 15) * ld + ((lane < 16) ? 0 : 8);
    v16h a;
#pragma unroll
    for (int e = 0; e < 8; ++e) a[e] = (half_t)r0[e];
#pragma unroll
    for (int e = 0; e < 8; ++e) a[8 + e] = (half_t)r0[16 + e];
    return a;
}

// B fragment: 32x16 f16 from an N-major staging buffer sT[n*ld + k].
__device__ __forceinline__ v16h load_b_frag(const half_t* tileT, int ld) {
    int lane = threadIdx.x & 31;
    const half_t* p = tileT + (lane & 15) * ld + ((lane < 16) ? 0 : 8);
    v16h b;
#pragma unroll
    for (int e = 0; e < 8; ++e) b[e] = p[e];
#pragma unroll
    for (int e = 0; e < 8; ++e) b[8 + e] = p[16 + e];
    return b;
}

// ------------- async global->LDS gather (per-lane, ASYNCcnt) -------------
__device__ __forceinline__ void async_copy_f32(void* lds_dst, const float* gsrc) {
    unsigned dst = (unsigned)(uintptr_t)lds_dst;            // LDS byte offset
    unsigned long long src = (unsigned long long)(uintptr_t)gsrc;
    asm volatile("global_load_async_to_lds_b32 %0, %1, off"
                 :: "v"(dst), "v"(src) : "memory");
}
__device__ __forceinline__ void wait_async0() {
    asm volatile("s_wait_asynccnt 0x0" ::: "memory");
}

// ------------- TDM: 1-D contiguous tensor tile -> LDS (TENSORcnt) --------
// D# per ISA 8.3/8.4: group0 {count=1, lds_addr, global_addr, type=2},
// group1 {data_size=4B, tensor_dim0, tensor_dim1=1, tile_dim0, stride}.
// This toolchain's builtin takes 6 args (g0, g1, g2, g3, g4, cpol).
__device__ __forceinline__ void tdm_load_1d_f32(unsigned lds_addr,
                                                const void* gaddr,
                                                unsigned nelem) {
    unsigned long long ga = (unsigned long long)(uintptr_t)gaddr;
    v4u g0 = {};
    g0[0] = 1u;                                            // count=1 (valid, user)
    g0[1] = lds_addr;                                      // LDS byte address
    g0[2] = (unsigned)(ga & 0xffffffffu);                  // global_addr[31:0]
    g0[3] = (unsigned)((ga >> 32) & 0x1ffffffu) | (2u << 30); // ga[56:32] | type=2
    v8i g1 = {};
    g1[0] = (2 << 16);                                     // data_size = 4 bytes
    g1[1] = (int)((nelem & 0xffffu) << 16);                // tensor_dim0[15:0]
    g1[2] = (int)((nelem >> 16) & 0xffffu) | (1 << 16);    // tensor_dim0[31:16], tensor_dim1=1
    g1[3] = (int)((nelem & 0xffffu) << 16);                // tile_dim0
    g1[4] = 0;                                             // tile_dim1/2 unused
    g1[5] = (int)nelem;                                    // tensor_dim0_stride[31:0]
    v4i g2 = {}; v4i g3 = {};
    v8i g4 = {};
    __builtin_amdgcn_tensor_load_to_lds(g0, g1, g2, g3, g4, 0);
}

// ---------------- direction index mapping (k=0..3) ----------------
__device__ __forceinline__ int dir_map(int k, int t) {
    int l = (k & 2) ? (LL - 1 - t) : t;
    if (k & 1) { int h = l & (HH - 1); int w = l >> 7; l = h * WW + w; }
    return l;
}

__device__ __forceinline__ float softplus_f(float v) {
    return (v > 20.f) ? v : log1pf(__expf(v));
}
__device__ __forceinline__ float silu_f(float v) {
    return v / (1.f + __expf(-v));
}

// ---------------- K0: build fused projection weights E_k (80x48, f16) ----
__global__ void k_build_E(const float* __restrict__ xpw,
                          const float* __restrict__ dtw,
                          half_t* __restrict__ Eh) {
    int k = blockIdx.x;
    for (int i = threadIdx.x; i < EM * EKP; i += blockDim.x) {
        int row = i / EKP, j = i % EKP;
        float v = 0.f;
        if (j < DI) {
            if (row < DI) {
                for (int r = 0; r < RNK; ++r)
                    v += dtw[(k*DI + row)*RNK + r] * xpw[(k*38 + r)*DI + j];
            } else {
                v = xpw[(k*38 + (row - DI + RNK))*DI + j];
            }
        }
        Eh[(size_t)k*EM*EKP + i] = (half_t)v;
    }
}

// ---------------- K1: in_proj GEMM (65536x96 @ 96x96^T) + split ----------
// A tile (64x96 f32, contiguous 24KB) staged by the Tensor Data Mover.
__global__ __launch_bounds__(256) void k_inproj(const float* __restrict__ x,
                                                const float* __restrict__ wip,
                                                float* __restrict__ xcp,
                                                float* __restrict__ zbuf) {
    __shared__ float  sA[64 * DM];    // f32 activation tile (TDM destination)
    __shared__ half_t sW[DM * DM];    // N-major: sW[n*96 + k] == wip[n][k]
    const int tid = threadIdx.x;
    const long rowbase = (long)blockIdx.x * 64;

    if ((tid >> 5) == 0) {            // wave 0 issues the DMA
        tdm_load_1d_f32((unsigned)(uintptr_t)&sA[0], x + rowbase * DM, 64 * DM);
        __builtin_amdgcn_s_wait_tensorcnt(0);
    }
    for (int i = tid; i < DM * DM; i += 256) sW[i] = (half_t)wip[i];
    __syncthreads();

    const int wave = tid >> 5, lane = tid & 31;
    const int mt = wave >> 1, nh = wave & 1;          // 4 M-tiles x 2 N-halves
    const int drow = (lane < 16) ? 0 : 8, dcol = lane & 15;
    const int bi = (int)(rowbase / LL);

    v16h a0 = load_a_frag_f32(&sA[(mt*16)*DM +  0], DM);
    v16h a1 = load_a_frag_f32(&sA[(mt*16)*DM + 32], DM);
    v16h a2 = load_a_frag_f32(&sA[(mt*16)*DM + 64], DM);

    for (int ntl = 0; ntl < 3; ++ntl) {
        int nt = nh*3 + ntl;
        v8f acc = {};
        acc = wmma_f16(a0, load_b_frag(&sW[(nt*16)*DM +  0], DM), acc);
        acc = wmma_f16(a1, load_b_frag(&sW[(nt*16)*DM + 32], DM), acc);
        acc = wmma_f16(a2, load_b_frag(&sW[(nt*16)*DM + 64], DM), acc);
        int col = nt*16 + dcol;
#pragma unroll
        for (int r = 0; r < 8; ++r) {
            long row = rowbase + mt*16 + drow + r;
            long hw  = row % LL;
            float v = acc[r];
            if (col < DI) xcp[((long)bi*DI + col)*LL + hw] = v;     // NCHW for conv
            else          zbuf[row*DI + (col - DI)] = v;            // gate
        }
    }
}

// ---------------- K2: depthwise 3x3 conv + bias + SiLU -------------------
__global__ void k_conv(const float* __restrict__ xcp,
                       const float* __restrict__ cw,
                       const float* __restrict__ cb,
                       float* __restrict__ xcv,
                       half_t* __restrict__ xcvh) {
    long tid = (long)blockIdx.x * blockDim.x + threadIdx.x;
    if (tid >= (long)BB * DI * LL) return;
    int hw = (int)(tid % LL);
    int d  = (int)((tid / LL) % DI);
    int b  = (int)(tid / ((long)LL * DI));
    int h = hw >> 7, w = hw & (WW - 1);
    const float* base = xcp + ((long)b*DI + d)*LL;
    float acc = cb[d];
#pragma unroll
    for (int kh = -1; kh <= 1; ++kh) {
        int hy = h + kh;
        if (hy < 0 || hy >= HH) continue;
#pragma unroll
        for (int kw = -1; kw <= 1; ++kw) {
            int wx = w + kw;
            if (wx < 0 || wx >= WW) continue;
            acc += cw[d*9 + (kh+1)*3 + (kw+1)] * base[hy*WW + wx];
        }
    }
    float s = silu_f(acc);
    xcv[tid]  = s;
    xcvh[tid] = (half_t)s;
}

// ---------------- K3: fused x_proj+dt_proj GEMM (E_k @ X) ----------------
__global__ __launch_bounds__(256) void k_xproj(const half_t* __restrict__ Eh,
                                               const half_t* __restrict__ xcvh,
                                               float* __restrict__ dR,
                                               float* __restrict__ Bsb,
                                               float* __restrict__ Csb) {
    __shared__ half_t sE[EM * EKP];     // A: row-major [80][64]
    __shared__ half_t sXt[128 * EKP];   // B staging, N-major: [col][kk]
    const int bid = blockIdx.x;
    const int ct = bid & 127;
    const int k  = (bid >> 7) & 3;
    const int bi = bid >> 9;
    const int colbase = ct * 128;
    const int tid = threadIdx.x;

    for (int i = tid; i < EM * EKP; i += 256) sE[i] = Eh[(size_t)k*EM*EKP + i];
    for (int i = tid; i < 128 * EKP; i += 256) {
        int col = i & 127, kk = i >> 7;
        half_t v = (half_t)0.f;
        if (kk < DI) v = xcvh[((long)bi*DI + kk)*LL + colbase + col];
        sXt[col*EKP + kk] = v;
    }
    __syncthreads();

    const int wave = tid >> 5, lane = tid & 31;
    const int nt = wave;
    const int drow = (lane < 16) ? 0 : 8, dcol = lane & 15;
    v16h b0 = load_b_frag(&sXt[(nt*16)*EKP +  0], EKP);
    v16h b1 = load_b_frag(&sXt[(nt*16)*EKP + 32], EKP);
    const long gb = (long)(bi*NK + k);
    const int l = colbase + nt*16 + dcol;

    for (int mt = 0; mt < 5; ++mt) {
        v8f acc = {};
        acc = wmma_f16(load_a_frag(&sE[(mt*16)*EKP +  0], EKP), b0, acc);
        acc = wmma_f16(load_a_frag(&sE[(mt*16)*EKP + 32], EKP), b1, acc);
#pragma unroll
        for (int r = 0; r < 8; ++r) {
            int row = mt*16 + drow + r;
            float v = acc[r];
            if (row < DI)                 dR [(gb*DI + row)*LL + l] = v;
            else if (row < DI + DSTATE)   Bsb[(gb*DSTATE + (row-DI))*LL + l] = v;
            else                          Csb[(gb*DSTATE + (row-DI-DSTATE))*LL + l] = v;
        }
    }
}

// ---------------- K4: chunked scan pass 1 (async-LDS gather staging) -----
__global__ __launch_bounds__(256) void k_scan1(const float* __restrict__ dR,
                                               const float* __restrict__ xcv,
                                               const float* __restrict__ Bsb,
                                               const float* __restrict__ Alog,
                                               const float* __restrict__ dtb,
                                               float* __restrict__ Pb,
                                               float* __restrict__ Sb) {
    __shared__ float sd[16*DI], sx[16*DI], sB[16*DSTATE];
    const int bid = blockIdx.x;
    const int c  = bid & (NC - 1);
    const int k  = (bid >> 7) & 3;
    const int bi = bid >> 9;
    const int tid = threadIdx.x;
    const int nn = tid & 15, d0 = tid >> 4;

    float Av[3], bia[3], P[3], S[3];
#pragma unroll
    for (int p = 0; p < 3; ++p) {
        int d = d0 + 16*p;
        Av[p]  = -__expf(Alog[(k*DI + d)*DSTATE + nn]);
        bia[p] = dtb[k*DI + d];
        P[p] = 1.f; S[p] = 0.f;
    }
    const long gb = (long)(bi*NK + k);
    const int tch = c * CH;

    for (int st = 0; st < CH; st += 16) {
        __syncthreads();
        for (int i = tid; i < 112 * 16; i += 256) {   // 7 asyncs/thread
            int j = i >> 4, s = i & 15;
            int l = dir_map(k, tch + st + s);
            if (j < DI)
                async_copy_f32(&sd[s*DI + j],              &dR [(gb*DI + j)*LL + l]);
            else if (j < 2*DI)
                async_copy_f32(&sx[s*DI + (j-DI)],         &xcv[((long)bi*DI + (j-DI))*LL + l]);
            else
                async_copy_f32(&sB[s*DSTATE + (j-2*DI)],   &Bsb[(gb*DSTATE + (j-2*DI))*LL + l]);
        }
        wait_async0();
        __syncthreads();
        for (int s = 0; s < 16; ++s) {
            float bv = sB[s*DSTATE + nn];
#pragma unroll
            for (int p = 0; p < 3; ++p) {
                int d = d0 + 16*p;
                float delta = softplus_f(sd[s*DI + d] + bia[p]);
                float a  = __expf(delta * Av[p]);
                float bu = delta * sx[s*DI + d] * bv;
                P[p] *= a;
                S[p] = a*S[p] + bu;
            }
        }
    }
#pragma unroll
    for (int p = 0; p < 3; ++p) {
        int d = d0 + 16*p;
        long idx = ((gb*NC + c)*DI + d)*DSTATE + nn;
        Pb[idx] = P[p]; Sb[idx] = S[p];
    }
}

// ---------------- K5: serial combine over chunk summaries ----------------
__global__ void k_prefix(const float* __restrict__ Pb,
                         const float* __restrict__ Sb,
                         float* __restrict__ Hp) {
    const int bid = blockIdx.x;          // bi*NK + k
    const int tid = threadIdx.x;         // 768 = d*16 + n
    float h = 0.f;
    const long base = (long)bid * NC * DI * DSTATE;
    for (int c = 0; c < NC; ++c) {
        long idx = base + (long)c * DI * DSTATE + tid;
        Hp[idx] = h;
        h = Pb[idx]*h + Sb[idx];
    }
}

// ---------------- K6: chunked scan pass 3 (seeded rescan + y output) -----
__global__ __launch_bounds__(256) void k_scan2(const float* __restrict__ dR,
                                               const float* __restrict__ xcv,
                                               const float* __restrict__ Bsb,
                                               const float* __restrict__ Csb,
                                               const float* __restrict__ Alog,
                                               const float* __restrict__ dtb,
                                               const float* __restrict__ Dsw,
                                               const float* __restrict__ Hp,
                                               float* __restrict__ ysum) {
    __shared__ float sd[16*DI], sx[16*DI], sB[16*DSTATE], sC[16*DSTATE];
    const int bid = blockIdx.x;
    const int c  = bid & (NC - 1);
    const int k  = (bid >> 7) & 3;
    const int bi = bid >> 9;
    const int tid = threadIdx.x;
    const int nn = tid & 15, d0 = tid >> 4;

    float Av[3], bia[3], Dsv[3], h[3];
    const long gb = (long)(bi*NK + k);
#pragma unroll
    for (int p = 0; p < 3; ++p) {
        int d = d0 + 16*p;
        Av[p]  = -__expf(Alog[(k*DI + d)*DSTATE + nn]);
        bia[p] = dtb[k*DI + d];
        Dsv[p] = Dsw[k*DI + d];
        h[p] = Hp[((gb*NC + c)*DI + d)*DSTATE + nn];
    }
    const int tch = c * CH;

    for (int st = 0; st < CH; st += 16) {
        __syncthreads();
        for (int i = tid; i < 128 * 16; i += 256) {   // 8 asyncs/thread
            int j = i >> 4, s = i & 15;
            int l = dir_map(k, tch + st + s);
            if (j < DI)
                async_copy_f32(&sd[s*DI + j],            &dR [(gb*DI + j)*LL + l]);
            else if (j < 2*DI)
                async_copy_f32(&sx[s*DI + (j-DI)],       &xcv[((long)bi*DI + (j-DI))*LL + l]);
            else if (j < 2*DI+DSTATE)
                async_copy_f32(&sB[s*DSTATE + (j-2*DI)], &Bsb[(gb*DSTATE + (j-2*DI))*LL + l]);
            else
                async_copy_f32(&sC[s*DSTATE + (j-2*DI-DSTATE)],
                               &Csb[(gb*DSTATE + (j-2*DI-DSTATE))*LL + l]);
        }
        wait_async0();
        __syncthreads();
        for (int s = 0; s < 16; ++s) {
            int l = dir_map(k, tch + st + s);
            float bv = sB[s*DSTATE + nn];
            float cv = sC[s*DSTATE + nn];
#pragma unroll
            for (int p = 0; p < 3; ++p) {
                int d = d0 + 16*p;
                float xv = sx[s*DI + d];
                float delta = softplus_f(sd[s*DI + d] + bia[p]);
                float a  = __expf(delta * Av[p]);
                float bu = delta * xv * bv;
                h[p] = a*h[p] + bu;
                float part = h[p] * cv;
                part += __shfl_xor(part, 1, 16);
                part += __shfl_xor(part, 2, 16);
                part += __shfl_xor(part, 4, 16);
                part += __shfl_xor(part, 8, 16);
                if (nn == 0)
                    atomicAdd(&ysum[((long)bi*LL + l)*DI + d], part + Dsv[p]*xv);
            }
        }
    }
}

// ---------------- K7: LayerNorm + SiLU gate + out_proj WMMA GEMM ---------
__global__ __launch_bounds__(256) void k_out(const float* __restrict__ ysum,
                                             const float* __restrict__ zbuf,
                                             const float* __restrict__ lng,
                                             const float* __restrict__ lnb,
                                             const float* __restrict__ wo,
                                             float* __restrict__ out) {
    __shared__ half_t sAct[128 * EKP];   // A: [row][64] (padded K)
    __shared__ half_t sWoT[DM * EKP];    // B staging, N-major: [n][kk]
    const int tid = threadIdx.x;
    const long rowbase = (long)blockIdx.x * 128;

    for (int i = tid; i < DM * EKP; i += 256) {
        int n = i >> 6, kk = i & 63;
        sWoT[i] = (kk < DI) ? (half_t)wo[n*DI + kk] : (half_t)0.f;
    }
    if (tid < 128) {
        long row = rowbase + tid;
        const float* yr = ysum + row*DI;
        const float* zr = zbuf + row*DI;
        float s = 0.f, sq = 0.f;
        for (int c2 = 0; c2 < DI; ++c2) { float v = yr[c2]; s += v; sq += v*v; }
        float mu   = s * (1.f / DI);
        float var  = sq * (1.f / DI) - mu*mu;
        float rstd = rsqrtf(var + 1e-5f);
        for (int c2 = 0; c2 < DI; ++c2) {
            float yn = (yr[c2] - mu) * rstd * lng[c2] + lnb[c2];
            sAct[tid*EKP + c2] = (half_t)(yn * silu_f(zr[c2]));
        }
        for (int c2 = DI; c2 < EKP; ++c2) sAct[tid*EKP + c2] = (half_t)0.f;
    }
    __syncthreads();

    const int wave = tid >> 5, lane = tid & 31;
    const int drow = (lane < 16) ? 0 : 8, dcol = lane & 15;
    v16h a0 = load_a_frag(&sAct[(wave*16)*EKP +  0], EKP);
    v16h a1 = load_a_frag(&sAct[(wave*16)*EKP + 32], EKP);
    for (int nt = 0; nt < 6; ++nt) {
        v8f acc = {};
        acc = wmma_f16(a0, load_b_frag(&sWoT[(nt*16)*EKP +  0], EKP), acc);
        acc = wmma_f16(a1, load_b_frag(&sWoT[(nt*16)*EKP + 32], EKP), acc);
#pragma unroll
        for (int r = 0; r < 8; ++r) {
            long row = rowbase + wave*16 + drow + r;
            out[row*DM + nt*16 + dcol] = acc[r];
        }
    }
}

// ---------------- launch ----------------
extern "C" void kernel_launch(void* const* d_in, const int* in_sizes, int n_in,
                              void* d_out, int out_size, void* d_ws, size_t ws_size,
                              hipStream_t stream) {
    const float* x    = (const float*)d_in[0];
    const float* wip  = (const float*)d_in[1];
    const float* cw   = (const float*)d_in[2];
    const float* cb   = (const float*)d_in[3];
    const float* xpw  = (const float*)d_in[4];
    const float* dtw  = (const float*)d_in[5];
    const float* dtb  = (const float*)d_in[6];
    const float* Alog = (const float*)d_in[7];
    const float* Dsw  = (const float*)d_in[8];
    const float* lng  = (const float*)d_in[9];
    const float* lnb  = (const float*)d_in[10];
    const float* wo   = (const float*)d_in[11];

    const size_t N_BDL = (size_t)BB * DI * LL;
    const size_t N_Z   = (size_t)BB * LL * DI;
    const size_t N_DR  = (size_t)BB * NK * DI * LL;
    const size_t N_BS  = (size_t)BB * NK * DSTATE * LL;
    const size_t N_PS  = (size_t)BB * NK * NC * DI * DSTATE;

    char* base = (char*)d_ws; size_t off = 0;
    auto take = [&](size_t bytes) -> void* {
        void* p = base + off; off = (off + bytes + 255) & ~(size_t)255; return p;
    };
    float*  xcp  = (float*) take(N_BDL * 4);
    float*  xcv  = (float*) take(N_BDL * 4);
    half_t* xcvh = (half_t*)take(N_BDL * 2);
    float*  zbuf = (float*) take(N_Z   * 4);
    half_t* Eh   = (half_t*)take((size_t)NK * EM * EKP * 2);
    float*  dR   = (float*) take(N_DR  * 4);
    float*  Bsb  = (float*) take(N_BS  * 4);
    float*  Csb  = (float*) take(N_BS  * 4);
    float*  Pb   = (float*) take(N_PS  * 4);
    float*  Sb   = (float*) take(N_PS  * 4);
    float*  Hp   = (float*) take(N_PS  * 4);
    float*  ysum = (float*) take(N_Z   * 4);

    k_build_E<<<NK, 256, 0, stream>>>(xpw, dtw, Eh);
    k_inproj <<<(BB*LL)/64, 256, 0, stream>>>(x, wip, xcp, zbuf);
    k_conv   <<<(unsigned)((N_BDL + 255)/256), 256, 0, stream>>>(xcp, cw, cb, xcv, xcvh);
    k_xproj  <<<BB*NK*(LL/128), 256, 0, stream>>>(Eh, xcvh, dR, Bsb, Csb);
    k_scan1  <<<BB*NK*NC, 256, 0, stream>>>(dR, xcv, Bsb, Alog, dtb, Pb, Sb);
    k_prefix <<<BB*NK, DI*DSTATE, 0, stream>>>(Pb, Sb, Hp);
    (void)hipMemsetAsync(ysum, 0, N_Z * 4, stream);
    k_scan2  <<<BB*NK*NC, 256, 0, stream>>>(dR, xcv, Bsb, Csb, Alog, dtb, Dsw, Hp, ysum);
    k_out    <<<(BB*LL)/128, 256, 0, stream>>>(ysum, zbuf, lng, lnb, wo, (float*)d_out);
}
